// GVPConv_9663676416046
// MI455X (gfx1250) — matrix operand order repaired and checked by reading, not compiled, not run on hardware
//
#include <hip/hip_runtime.h>
#include <math.h>

// ---------------- CDNA5 WMMA types ----------------
typedef __attribute__((ext_vector_type(16))) _Float16 v16h;
typedef __attribute__((ext_vector_type(8)))  float    v8f;
typedef __attribute__((ext_vector_type(8)))  _Float16 h8;
typedef __attribute__((ext_vector_type(4)))  unsigned int u32x4;

union FragH { h8 h[2]; v16h v; u32x4 q[2]; };

#define EPW 64      // edges per workgroup (2 nodes x 32 neighbors)
#define KNB 32      // neighbors per node

__device__ __forceinline__ float sigm(float x) { return 1.0f / (1.0f + __expf(-x)); }

// ---------------- weight pre-pack kernel ----------------
// Packs w[rows][F] (row-major f32) into per-lane B fragments for
// v_wmma_f32_16x16x32_f16.  Fragment (nT,kT): lane holds column o = nT*16 +
// lane%16; kb = kT*32 + (lane<16 ? 0 : 8); elements 0..7 hold K = kb..kb+7,
// elements 8..15 hold K = kb+16..kb+23.  K >= F is zero-padded.
__global__ void pack_wmma_b(const float* __restrict__ w, int F, int kTiles,
                            u32x4* __restrict__ dst)
{
    const int frag = blockIdx.x;               // frag = nT*kTiles + kT
    const int nT   = frag / kTiles;
    const int kt   = frag - nT * kTiles;
    const int lane = threadIdx.x;              // 0..31
    const int o    = nT * 16 + (lane & 15);
    const int kb   = kt * 32 + ((lane >> 4) << 3);
    union { _Float16 h[16]; u32x4 q[2]; } u;
#pragma unroll
    for (int i = 0; i < 16; ++i) {
        const int k = kb + (i < 8 ? i : i + 8);
        u.h[i] = (_Float16)((k < F) ? w[o * F + k] : 0.0f);
    }
    u32x4* d = dst + (frag * 32 + lane) * 2;
    d[0] = u.q[0];
    d[1] = u.q[1];
}

// ---------------- main kernel ----------------
struct Params {
    const float* s;        // [4096][128]
    const float* v;        // [4096][16][3]
    const float* edge_s;   // [4096][32][32]
    const float* edge_v;   // [4096][32][3]
    const int*   edge_index; // [4096][32]
    const float *wh1, *wsb1, *wv1, *wsvb1;
    const float *wh2, *wsb2, *wv2, *wsvb2;
    const float *wh3, *wsb3, *wv3, *wsvb3;
    const u32x4 *Bw1, *Bw2, *Bw3, *G1, *G2, *G3;
    float* out_s;          // [4096][128]
    float* out_v;          // [4096][48]
};

// Generic GVP layer.  SI: scalar in, H: hidden vec dim, VI: vec in, LDF:
// padded feature row stride (f16), KT: k-tiles (LDF/32), VHS/MVS: f32 row
// strides for vh / mv-in buffers.  PADH = LDF - SI (vn + zero pad region).
template<int SI, int H, int VI, int LDF, int KT, int VHS, int MVS, int PADH, bool ACT>
__device__ void gvp_layer(
    _Float16* msBuf, _Float16* sigBuf, float* soBuf, float* vhBuf,
    const float* mvIn, float* mvOut, float* gateBuf,
    float* whLds, float* wvLds,
    const float* __restrict__ wh, const float* __restrict__ wsb,
    const float* __restrict__ wv, const float* __restrict__ wsvb,
    const u32x4* __restrict__ Bw, const u32x4* __restrict__ Gw)
{
    const int tid  = threadIdx.x;
    const int lane = tid & 31;
    const int wave = tid >> 5;

    // preload small weights (wh: [H][VI], wv: [16][H]) into LDS
    for (int i = tid; i < H * VI; i += 256) whLds[i] = wh[i];
    for (int i = tid; i < 16 * H; i += 256) wvLds[i] = wv[i];
    __syncthreads();

    // ---- S1: vh = v^T * wh^T per edge; vn = ||vh||; zero K-padding ----
    for (int pidx = tid; pidx < EPW * PADH; pidx += 256) {
        const int e  = pidx / PADH;
        const int hh = pidx % PADH;
        if (hh < H) {
            const float* mv = mvIn + e * MVS;
            float a0 = 0.f, a1 = 0.f, a2 = 0.f;
#pragma unroll 4
            for (int i = 0; i < VI; ++i) {
                const float  w  = whLds[hh * VI + i];
                const float4 m4 = *(const float4*)(mv + i * 4);
                a0 = fmaf(w, m4.x, a0);
                a1 = fmaf(w, m4.y, a1);
                a2 = fmaf(w, m4.z, a2);
            }
            float* vp = vhBuf + e * VHS + hh * 4;
            vp[0] = a0; vp[1] = a1; vp[2] = a2;
            const float ss = a0 * a0 + a1 * a1 + a2 * a2;
            msBuf[e * LDF + SI + hh] = (_Float16)sqrtf(fmaxf(ss, 1e-8f));
        } else {
            msBuf[e * LDF + SI + hh] = (_Float16)0.0f;  // K padding
        }
    }
    __syncthreads();

    // ---- S2: so = [64 x LDF] * [LDF x 128] via WMMA; wave = N-tile ----
    {
        const int   n     = lane & 15;
        const int   koff  = (lane >> 4) << 3;  // K sub-offset for A frags
        const int   rbase = (lane >> 4) << 3;  // C/D row base
        const float bias  = wsb[wave * 16 + n];
#pragma unroll
        for (int mt = 0; mt < 4; ++mt) {
            v8f acc = {};
#pragma unroll
            for (int kt = 0; kt < KT; ++kt) {
                FragH A, B;
                const _Float16* ap = msBuf + (mt * 16 + n) * LDF + kt * 32 + koff;
                A.h[0] = *(const h8*)(ap);
                A.h[1] = *(const h8*)(ap + 16);
                const u32x4* bp = Bw + ((wave * KT + kt) * 32 + lane) * 2;
                B.q[0] = bp[0];
                B.q[1] = bp[1];
                acc = __builtin_amdgcn_wmma_f32_16x16x32_f16(
                    false, A.v, false, B.v, (short)0, acc, false, false);
            }
#pragma unroll
            for (int r = 0; r < 8; ++r)
                soBuf[(mt * 16 + rbase + r) * 128 + wave * 16 + n] = acc[r] + bias;
        }
    }
    __syncthreads();

    // ---- S3: gate_in = sigmoid(so) (or so), next-layer scalars = relu(so) ----
    for (int pidx = tid; pidx < EPW * 128; pidx += 256) {
        const float x = soBuf[pidx];
        sigBuf[pidx]  = (_Float16)(ACT ? sigm(x) : x);
        if (ACT) {
            const int e = pidx >> 7, o = pidx & 127;
            msBuf[e * 160 + o] = (_Float16)fmaxf(x, 0.0f);  // next layer LDF=160
        }
    }
    __syncthreads();

    // ---- S4: gate GEMM (waves 0-3, WMMA) | vo = vh*wv^T (waves 4-7, VALU) ----
    if (wave < 4) {
        const int   n     = lane & 15;
        const int   koff  = (lane >> 4) << 3;
        const int   rbase = (lane >> 4) << 3;
        const float bias  = wsvb[n];
        v8f acc = {};
#pragma unroll
        for (int kt = 0; kt < 4; ++kt) {
            FragH A, B;
            const _Float16* ap = sigBuf + (wave * 16 + n) * 128 + kt * 32 + koff;
            A.h[0] = *(const h8*)(ap);
            A.h[1] = *(const h8*)(ap + 16);
            const u32x4* bp = Gw + (kt * 32 + lane) * 2;
            B.q[0] = bp[0];
            B.q[1] = bp[1];
            acc = __builtin_amdgcn_wmma_f32_16x16x32_f16(
                false, A.v, false, B.v, (short)0, acc, false, false);
        }
#pragma unroll
        for (int r = 0; r < 8; ++r)
            gateBuf[(wave * 16 + rbase + r) * 16 + n] = sigm(acc[r] + bias);
    } else {
        const int t2 = tid - 128;
        for (int j = t2; j < EPW * 16; j += 128) {
            const int e = j >> 4, o = j & 15;
            float a0 = 0.f, a1 = 0.f, a2 = 0.f;
            for (int hh = 0; hh < H; ++hh) {
                const float  w  = wvLds[o * H + hh];
                const float4 m4 = *(const float4*)(vhBuf + e * VHS + hh * 4);
                a0 = fmaf(w, m4.x, a0);
                a1 = fmaf(w, m4.y, a1);
                a2 = fmaf(w, m4.z, a2);
            }
            float* op = mvOut + e * 64 + o * 4;
            op[0] = a0; op[1] = a1; op[2] = a2;
        }
    }
    __syncthreads();

    // ---- S5: vo *= sigmoid(gate) ----
    for (int j = tid; j < EPW * 16; j += 256) {
        const int e = j >> 4, o = j & 15;
        const float gg = gateBuf[e * 16 + o];
        float* op = mvOut + e * 64 + o * 4;
        op[0] *= gg; op[1] *= gg; op[2] *= gg;
    }
    __syncthreads();
}

__global__ __launch_bounds__(256)
void gvpconv_kernel(Params p)
{
    __shared__ __align__(16) _Float16 msBuf[EPW * 352];  // 45056 B (f16 features)
    __shared__ __align__(16) float    soBuf[EPW * 128];  // 32768 B
    __shared__ __align__(16) float    vhBuf[EPW * 132];  // 33792 B (vec4 per h)
    __shared__ __align__(16) float    mvA[EPW * 132];    // 33792 B (vec4 per vi)
    __shared__ __align__(16) float    mvB[EPW * 64];     // 16384 B
    __shared__ float gateBuf[EPW * 16];                  //  4096 B
    __shared__ float whLds[33 * 33];
    __shared__ float wvLds[16 * 33];

    const int tid  = threadIdx.x;
    const int lane = tid & 31;
    const int wave = tid >> 5;
    const int g0   = blockIdx.x * 2;          // 2 nodes per workgroup
    _Float16* sigBuf = msBuf + EPW * 160;     // overlay in dead msBuf region

    // ---- gather + concat: ms = [xi_s | xj_s | edge_s], mv = [xi_v;xj_v;edge_v]
    for (int e8 = 0; e8 < 8; ++e8) {
        const int e   = wave * 8 + e8;
        const int g   = g0 + (e >> 5);
        const int b   = g >> 11;
        const int k   = e & 31;
        const int idx = p.edge_index[g * KNB + k];
        const int gn  = (b << 11) + idx;

        _Float16* row = msBuf + e * 352;
        const float4 xi = *(const float4*)(p.s + g  * 128 + lane * 4);
        const float4 xj = *(const float4*)(p.s + gn * 128 + lane * 4);
        row[lane * 4 + 0] = (_Float16)xi.x;  row[lane * 4 + 1] = (_Float16)xi.y;
        row[lane * 4 + 2] = (_Float16)xi.z;  row[lane * 4 + 3] = (_Float16)xi.w;
        row[128 + lane * 4 + 0] = (_Float16)xj.x;
        row[128 + lane * 4 + 1] = (_Float16)xj.y;
        row[128 + lane * 4 + 2] = (_Float16)xj.z;
        row[128 + lane * 4 + 3] = (_Float16)xj.w;
        row[256 + lane] = (_Float16)p.edge_s[(g * KNB + k) * 32 + lane];

        float* mvrow = mvA + e * 132;
        if (lane < 16) {
            const float* vp = p.v + g * 48 + lane * 3;
            mvrow[lane * 4 + 0] = vp[0];
            mvrow[lane * 4 + 1] = vp[1];
            mvrow[lane * 4 + 2] = vp[2];
        } else {
            const int    l2 = lane - 16;
            const float* vp = p.v + gn * 48 + l2 * 3;
            mvrow[(16 + l2) * 4 + 0] = vp[0];
            mvrow[(16 + l2) * 4 + 1] = vp[1];
            mvrow[(16 + l2) * 4 + 2] = vp[2];
        }
        if (lane == 0) {
            const float* ep = p.edge_v + (g * KNB + k) * 3;
            mvrow[128 + 0] = ep[0]; mvrow[128 + 1] = ep[1]; mvrow[128 + 2] = ep[2];
        }
    }
    __syncthreads();

    // ---- 3 GVP layers ----
    gvp_layer<288, 33, 33, 352, 11, 132, 132, 64, true >(
        msBuf, sigBuf, soBuf, vhBuf, mvA, mvB, gateBuf, whLds, wvLds,
        p.wh1, p.wsb1, p.wv1, p.wsvb1, p.Bw1, p.G1);
    gvp_layer<128, 16, 16, 160,  5,  64,  64, 32, true >(
        msBuf, sigBuf, soBuf, vhBuf, mvB, mvA, gateBuf, whLds, wvLds,
        p.wh2, p.wsb2, p.wv2, p.wsvb2, p.Bw2, p.G2);
    gvp_layer<128, 16, 16, 160,  5,  64,  64, 32, false>(
        msBuf, sigBuf, soBuf, vhBuf, mvA, mvB, gateBuf, whLds, wvLds,
        p.wh3, p.wsb3, p.wv3, p.wsvb3, p.Bw3, p.G3);

    // ---- aggregate: mean over K=32 neighbors + residual ----
    {
        const int j = tid >> 7, o = tid & 127;
        const int g = g0 + j;
        float acc = 0.f;
#pragma unroll
        for (int k = 0; k < 32; ++k) acc += soBuf[(j * 32 + k) * 128 + o];
        p.out_s[g * 128 + o] = p.s[g * 128 + o] + acc * 0.03125f;
    }
    if (tid < 96) {
        const int j = tid / 48, r = tid % 48;
        const int i = r / 3, t = r % 3;
        const int g = g0 + j;
        float acc = 0.f;
#pragma unroll
        for (int k = 0; k < 32; ++k) acc += mvB[(j * 32 + k) * 64 + i * 4 + t];
        p.out_v[g * 48 + r] = p.v[g * 48 + r] + acc * 0.03125f;
    }
}

// ---------------- host launch ----------------
extern "C" void kernel_launch(void* const* d_in, const int* in_sizes, int n_in,
                              void* d_out, int out_size, void* d_ws, size_t ws_size,
                              hipStream_t stream)
{
    (void)in_sizes; (void)n_in; (void)out_size; (void)ws_size;

    // scratch layout: packed f16 B-fragments (1024 B per 16x16x32 fragment)
    char* ws = (char*)d_ws;
    u32x4* Bw1 = (u32x4*)(ws + 0);        // 8 nT * 11 kT = 88 frags = 90112 B
    u32x4* Bw2 = (u32x4*)(ws + 90112);    // 8 * 5 = 40 frags = 40960 B
    u32x4* Bw3 = (u32x4*)(ws + 131072);   // 40960 B
    u32x4* G1  = (u32x4*)(ws + 172032);   // 1 * 4 = 4 frags = 4096 B
    u32x4* G2  = (u32x4*)(ws + 176128);
    u32x4* G3  = (u32x4*)(ws + 180224);

    pack_wmma_b<<<88, 32, 0, stream>>>((const float*)d_in[6],  321, 11, Bw1);
    pack_wmma_b<<<40, 32, 0, stream>>>((const float*)d_in[12], 144,  5, Bw2);
    pack_wmma_b<<<40, 32, 0, stream>>>((const float*)d_in[18], 144,  5, Bw3);
    pack_wmma_b<<< 4, 32, 0, stream>>>((const float*)d_in[9],  128,  4, G1);
    pack_wmma_b<<< 4, 32, 0, stream>>>((const float*)d_in[15], 128,  4, G2);
    pack_wmma_b<<< 4, 32, 0, stream>>>((const float*)d_in[21], 128,  4, G3);

    Params p;
    p.s          = (const float*)d_in[0];
    p.v          = (const float*)d_in[1];
    p.edge_s     = (const float*)d_in[2];
    p.edge_v     = (const float*)d_in[3];
    p.edge_index = (const int*)  d_in[4];
    p.wh1 = (const float*)d_in[5];  p.wsb1 = (const float*)d_in[7];
    p.wv1 = (const float*)d_in[8];  p.wsvb1 = (const float*)d_in[10];
    p.wh2 = (const float*)d_in[11]; p.wsb2 = (const float*)d_in[13];
    p.wv2 = (const float*)d_in[14]; p.wsvb2 = (const float*)d_in[16];
    p.wh3 = (const float*)d_in[17]; p.wsb3 = (const float*)d_in[19];
    p.wv3 = (const float*)d_in[20]; p.wsvb3 = (const float*)d_in[22];
    p.Bw1 = Bw1; p.Bw2 = Bw2; p.Bw3 = Bw3;
    p.G1 = G1;  p.G2 = G2;  p.G3 = G3;
    p.out_s = (float*)d_out;
    p.out_v = (float*)d_out + 2 * 2048 * 128;

    gvpconv_kernel<<<2048, 256, 0, stream>>>(p);
}
